// NaiveCustomLSTM_43748536877595
// MI455X (gfx1250) — matrix operand order repaired and checked by reading, not compiled, not run on hardware
//
#include <hip/hip_runtime.h>

typedef __attribute__((ext_vector_type(16))) _Float16 v16h;
typedef __attribute__((ext_vector_type(8)))  _Float16 v8h;
typedef __attribute__((ext_vector_type(8)))  float    v8f;
typedef __attribute__((ext_vector_type(4)))  float    v4f;

#define INPUT_SZ 16
#define HPF      32
#define NBLK     20            // hidden blocks (block-diagonal V)
#define HIDDEN   640
#define BATCH    128
#define SEQ      512
#define WU_TILES 160           // 2560/16 N-tiles of folded-U (K=16 padded to 32)
#define V_TILES  160           // 4 gates * 20 blocks * 2 n-halves

// ---------------------------------------------------------------------------
// Prep: fold FEAT_SEQ gather + U_MASK into a 16x2560 f16 weight, apply V_MASK
// (block-diagonal) to V, and store both pre-swizzled in the WMMA B-operand
// per-lane layout: tile = 32 lanes x 16 halves contiguous (lane l: N=l&15,
// K = (l>>4)*8 + j for j<8, 16 + (l>>4)*8 + (j-8) for j>=8). One thread per (tile, lane).
// ---------------------------------------------------------------------------
__global__ void prep_weights(const float* __restrict__ U_i, const float* __restrict__ V_i,
                             const float* __restrict__ U_f, const float* __restrict__ V_f,
                             const float* __restrict__ U_c, const float* __restrict__ V_c,
                             const float* __restrict__ U_o, const float* __restrict__ V_o,
                             _Float16* __restrict__ wu_sw, _Float16* __restrict__ v_sw) {
  int tid = blockIdx.x * blockDim.x + threadIdx.x;
  if (tid >= (WU_TILES + V_TILES) * 32) return;
  int tile = tid >> 5, lane = tid & 31;
  int n = lane & 15, lg = lane >> 4;
  const float* Us[4] = {U_i, U_f, U_c, U_o};
  const float* Vs[4] = {V_i, V_f, V_c, V_o};

  if (tile < WU_TILES) {
    int col = tile * 16 + n;             // column in 2560-wide Ucat
    int g   = col / HIDDEN;
    int hid = col % HIDDEN;
    int blk = hid / HPF;
    const float* U = Us[g];
    _Float16* dst = wu_sw + ((size_t)tile * 32 + lane) * 16;
    for (int j = 0; j < 16; ++j) {
      int K = (j < 8) ? (lg * 8 + j) : (16 + lg * 8 + (j - 8));   // padded K (>=16 -> zero)
      float v = 0.f;
      if (K < INPUT_SZ) {
        // sum over gather positions p with FEAT_SEQ[p]==K, masked by U_MASK
        for (int p = 0; p < 24; ++p) {
          int feat = (p < 16) ? p : (p - 16);
          int mblk = (p < 16) ? p : ((p - 16) & ~1);
          if (feat == K && mblk == blk) v += U[p * HIDDEN + hid];
        }
      }
      dst[j] = (_Float16)v;
    }
  } else {
    int vt   = tile - WU_TILES;          // (g*20 + blk)*2 + half
    int half = vt & 1;
    int rest = vt >> 1;
    int blk  = rest % NBLK;
    int g    = rest / NBLK;
    const float* V = Vs[g];
    _Float16* dst = v_sw + ((size_t)vt * 32 + lane) * 16;
    int colh = blk * HPF + half * 16 + n;
    for (int j = 0; j < 16; ++j) {
      int K = (j < 8) ? (lg * 8 + j) : (16 + lg * 8 + (j - 8));   // 0..31, row within diagonal block
      int r = blk * HPF + K;
      dst[j] = (_Float16)V[(size_t)r * HIDDEN + colh];
    }
  }
}

// Fast transcendentals: one v_exp_f32 + one v_rcp_f32 each, no IEEE div chain.
__device__ __forceinline__ float frcp(float x)  { return __builtin_amdgcn_rcpf(x); }
__device__ __forceinline__ float fsig(float x)  { return frcp(1.f + __expf(-x)); }
__device__ __forceinline__ float ftanh(float x) { float e = __expf(-2.f * x); return (1.f - e) * frcp(1.f + e); }

// ---------------------------------------------------------------------------
// Fused LSTM recurrence. Block-diagonal V means each (16-batch tile, 32-wide
// hidden block) is an independent size-32 LSTM. grid = 8 batch-tiles x 20
// blocks; 2 waves per workgroup, one per 16-column half. Per step per wave:
// 4 WMMAs for x@U' (bias in C) chained into 4 WMMAs for h@V_block.
// h is double-buffered in LDS -> one workgroup barrier per step.
// ---------------------------------------------------------------------------
__global__ void __launch_bounds__(64) lstm_rec(
    const float* __restrict__ x,
    const float* __restrict__ b_i, const float* __restrict__ b_f,
    const float* __restrict__ b_c, const float* __restrict__ b_o,
    const _Float16* __restrict__ wu_sw, const _Float16* __restrict__ v_sw,
    float* __restrict__ out) {
  __shared__ __align__(32) _Float16 hbuf[2][16 * 32];   // double-buffered h tile

  const int bt   = blockIdx.x & 7;
  const int blk  = blockIdx.x >> 3;
  const int half = threadIdx.x >> 5;   // wave id = n-half
  const int lane = threadIdx.x & 31;
  const int n    = lane & 15;          // B/C/D column within 16-wide half
  const int lg   = lane >> 4;
  const int m    = lane & 15;          // A-matrix row for this lane

  for (int i = threadIdx.x; i < 2 * 16 * 32; i += 64) hbuf[0][i] = (_Float16)0.f;

  // Resident B operands: 4 V-tiles + 4 U'-tiles for this (blk, half).
  v16h VB[4], UB[4];
#pragma unroll
  for (int g = 0; g < 4; ++g) {
    int vt = (g * NBLK + blk) * 2 + half;
    VB[g] = *(const v16h*)(v_sw + ((size_t)vt * 32 + lane) * 16);
    int ut = g * (HIDDEN / 16) + blk * 2 + half;   // (g*640 + blk*32 + half*16)/16
    UB[g] = *(const v16h*)(wu_sw + ((size_t)ut * 32 + lane) * 16);
  }
  const float* bs[4] = {b_i, b_f, b_c, b_o};
  const int colh = blk * HPF + half * 16 + n;
  v8f biasv[4];
#pragma unroll
  for (int g = 0; g < 4; ++g) {
    float bv = bs[g][colh];
#pragma unroll
    for (int r = 0; r < 8; ++r) biasv[g][r] = bv;
  }

  v8f c = {};   // cell state, f32, accumulator layout
  v8f hl = {};  // last h, for final-state output
  const size_t xrowbase = ((size_t)(bt * 16 + m) * SEQ) * INPUT_SZ + lg * 8;
  __syncthreads();

  for (int t = 0; t < SEQ; ++t) {
    const _Float16* hrd = hbuf[t & 1];
    _Float16*       hwr = hbuf[(t + 1) & 1];

    // A_h (16x32 f16): halves 0..7 -> K = lg*8+0..7, 8..15 -> K = 16+lg*8+0..7
    v16h ah;
    {
      v8h lo = *(const v8h*)(hrd + m * 32 + lg * 8);
      v8h hi = *(const v8h*)(hrd + m * 32 + 16 + lg * 8);
#pragma unroll
      for (int j = 0; j < 8; ++j) { ah[j] = lo[j]; ah[8 + j] = hi[j]; }
    }
    // A_x (16x32 f16, K=16 real + 16 zero pad)
    v16h ax = {};
    {
      const float* xp = x + xrowbase + (size_t)t * INPUT_SZ;
      v4f f0 = *(const v4f*)xp;
      v4f f1 = *(const v4f*)(xp + 4);
#pragma unroll
      for (int j = 0; j < 4; ++j) { ax[j] = (_Float16)f0[j]; ax[4 + j] = (_Float16)f1[j]; }
    }
    __builtin_prefetch(x + xrowbase + (size_t)(t + 1) * INPUT_SZ, 0, 3);

    v8f acc[4];
#pragma unroll
    for (int g = 0; g < 4; ++g) {
      acc[g] = __builtin_amdgcn_wmma_f32_16x16x32_f16(false, ax, false, UB[g],
                                                      (short)0, biasv[g], false, false);
      acc[g] = __builtin_amdgcn_wmma_f32_16x16x32_f16(false, ah, false, VB[g],
                                                      (short)0, acc[g], false, false);
    }

#pragma unroll
    for (int r = 0; r < 8; ++r) {
      float iv = fsig(acc[0][r]);
      float fv = fsig(acc[1][r]);
      float gv = ftanh(acc[2][r]);
      float ov = fsig(acc[3][r]);
      float cn = fv * c[r] + iv * gv;
      c[r] = cn;
      float hn = ov * ftanh(cn);
      hl[r] = hn;
      int mrow = r + 8 * lg;                       // accumulator-layout row
      hwr[mrow * 32 + half * 16 + n] = (_Float16)hn;
      int b = bt * 16 + mrow;
      out[((size_t)b * SEQ + t) * HIDDEN + colh] = hn;
    }
    __syncthreads();   // new h visible before next step reads hwr
  }

  // final (h_t, c_t) appended after hidden_seq
  float* outH = out + (size_t)BATCH * SEQ * HIDDEN;
  float* outC = outH + (size_t)BATCH * HIDDEN;
#pragma unroll
  for (int r = 0; r < 8; ++r) {
    int b = bt * 16 + r + 8 * lg;
    outH[(size_t)b * HIDDEN + colh] = hl[r];
    outC[(size_t)b * HIDDEN + colh] = c[r];
  }
}

extern "C" void kernel_launch(void* const* d_in, const int* in_sizes, int n_in,
                              void* d_out, int out_size, void* d_ws, size_t ws_size,
                              hipStream_t stream) {
  const float* x   = (const float*)d_in[0];
  const float* U_i = (const float*)d_in[1];
  const float* V_i = (const float*)d_in[2];
  const float* b_i = (const float*)d_in[3];
  const float* U_f = (const float*)d_in[4];
  const float* V_f = (const float*)d_in[5];
  const float* b_f = (const float*)d_in[6];
  const float* U_c = (const float*)d_in[7];
  const float* V_c = (const float*)d_in[8];
  const float* b_c = (const float*)d_in[9];
  const float* U_o = (const float*)d_in[10];
  const float* V_o = (const float*)d_in[11];
  const float* b_o = (const float*)d_in[12];

  _Float16* wu_sw = (_Float16*)d_ws;                       // 160 tiles * 1 KB
  _Float16* v_sw  = wu_sw + (size_t)WU_TILES * 32 * 16;    // 160 tiles * 1 KB

  prep_weights<<<(((WU_TILES + V_TILES) * 32) + 255) / 256, 256, 0, stream>>>(
      U_i, V_i, U_f, V_f, U_c, V_c, U_o, V_o, wu_sw, v_sw);

  lstm_rec<<<dim3(8 * NBLK), 64, 0, stream>>>(
      x, b_i, b_f, b_c, b_o, wu_sw, v_sw, (float*)d_out);
}
